// CoAttention_46042049413525
// MI455X (gfx1250) — matrix operand order repaired and checked by reading, not compiled, not run on hardware
//
#include <hip/hip_runtime.h>

typedef __attribute__((ext_vector_type(16))) __bf16 bf16x16;
typedef __attribute__((ext_vector_type(2)))  __bf16 bf16x2;
typedef __attribute__((ext_vector_type(8)))  float  v8f;
typedef __attribute__((ext_vector_type(2)))  float  f32x2;

#define BZ      32
#define RV_NUM  10
#define RV_LEN  128
#define IN_FEAT 300
#define HDIM    128
#define NREV    (BZ * RV_NUM)        // 320 reviews per side
#define ROWS    (NREV * RV_LEN)      // 40960 token rows per tensor
#define KTILE   32
#define NKT     10                   // K padded to 320
#define LDT     40                   // LDS row stride in bf16 elems (80B, 16B-aligned)

union FragAB { uint4 u[2]; bf16x16 v; };

// packed f32 -> bf16 conversion: lets the backend pick v_cvt_pk_bf16_f32
__device__ __forceinline__ unsigned pack_bf16x2(float a, float b) {
  f32x2 v; v.x = a; v.y = b;
  bf16x2 h = __builtin_convertvector(v, bf16x2);
  union { bf16x2 h; unsigned u; } p; p.h = h;
  return p.u;
}

// ---------------------------------------------------------------------------
// Kernel 1: fused FC + ReLU:  dst[m,n] = relu( src[m,:300] @ W[:,n] + bias[n] )
// Block = 128(M) x 128(N), 8 waves (wave32), 16 M-rows per wave.
// bf16 WMMA 16x16x32 f32-acc, double-buffered LDS, one barrier per K-chunk.
// grid = (320, 2) for seq_a / seq_b.
// ---------------------------------------------------------------------------
__global__ __launch_bounds__(256)
void fc_wmma_kernel(const float* __restrict__ seqA, const float* __restrict__ seqB,
                    const float* __restrict__ W,    const float* __restrict__ bias,
                    float* __restrict__ ta,         float* __restrict__ tb)
{
  __shared__ unsigned short ldsA[2][128 * LDT];  // [m 0..127][k 0..31] bf16
  __shared__ unsigned short ldsW[2][128 * LDT];  // W^T: [n 0..127][k 0..31] bf16

  const int tid  = threadIdx.x;
  const int lane = tid & 31;
  const int wave = tid >> 5;
  const int mrow = lane & 15;
  const int hi   = lane >> 4;

  const float* src = (blockIdx.y == 0) ? seqA : seqB;
  float*       dst = (blockIdx.y == 0) ? ta   : tb;
  const int rowBase = blockIdx.x * 128;

  float4 aReg[4];
  float  wReg[8][2];

  // ---- staging helpers (macros keep everything in registers) ----
#define LOAD_CHUNK(K0)                                                         \
  {                                                                            \
    _Pragma("unroll")                                                          \
    for (int it = 0; it < 4; ++it) {                                           \
      int i  = tid + it * 256;           /* 0..1023 */                         \
      int r  = i >> 3;                   /* 0..127  */                         \
      int c4 = (i & 7) * 4;              /* 0,4..28 */                         \
      int k  = (K0) + c4;                                                      \
      if (k < IN_FEAT)                                                         \
        aReg[it] = *(const float4*)(src + (size_t)(rowBase + r) * IN_FEAT + k);\
      else                                                                     \
        aReg[it] = make_float4(0.f, 0.f, 0.f, 0.f);                            \
    }                                                                          \
    _Pragma("unroll")                                                          \
    for (int it = 0; it < 8; ++it) {                                           \
      int i  = tid + it * 256;           /* 0..2047 */                         \
      int kp = i >> 7;                   /* 0..15   */                         \
      int n  = i & 127;                                                        \
      int k  = (K0) + 2 * kp;                                                  \
      wReg[it][0] = (k     < IN_FEAT) ? W[(size_t)k * HDIM + n]       : 0.f;   \
      wReg[it][1] = (k + 1 < IN_FEAT) ? W[(size_t)(k + 1) * HDIM + n] : 0.f;   \
    }                                                                          \
  }

#define STORE_CHUNK(BUF)                                                       \
  {                                                                            \
    _Pragma("unroll")                                                          \
    for (int it = 0; it < 4; ++it) {                                           \
      int i  = tid + it * 256;                                                 \
      int r  = i >> 3;                                                         \
      int c4 = (i & 7) * 4;                                                    \
      uint2 p;                                                                 \
      p.x = pack_bf16x2(aReg[it].x, aReg[it].y);                               \
      p.y = pack_bf16x2(aReg[it].z, aReg[it].w);                               \
      *(uint2*)&ldsA[BUF][r * LDT + c4] = p;                                   \
    }                                                                          \
    _Pragma("unroll")                                                          \
    for (int it = 0; it < 8; ++it) {                                           \
      int i  = tid + it * 256;                                                 \
      int kp = i >> 7;                                                         \
      int n  = i & 127;                                                        \
      *(unsigned*)&ldsW[BUF][n * LDT + 2 * kp] =                               \
          pack_bf16x2(wReg[it][0], wReg[it][1]);                               \
    }                                                                          \
  }

  v8f acc[8] = {};

  // prologue: stage chunk 0 into buffer 0
  LOAD_CHUNK(0)
  STORE_CHUNK(0)

  int buf = 0;
  for (int kk = 0; kk < NKT; ++kk) {
    __syncthreads();                       // buffer `buf` ready for all waves

    const bool hasNext = (kk + 1 < NKT);   // wave-uniform
    if (hasNext) LOAD_CHUNK((kk + 1) * KTILE)   // kick off next-chunk VMEM

    // A fragment: lane<16 -> K{0..7,16..23}, lane>=16 -> K{8..15,24..31}
    FragAB a;
    const unsigned short* ap = &ldsA[buf][(wave * 16 + mrow) * LDT];
    a.u[0] = *(const uint4*)(ap + hi * 8);
    a.u[1] = *(const uint4*)(ap + 16 + hi * 8);

    #pragma unroll
    for (int nt = 0; nt < 8; ++nt) {
      FragAB b;
      const unsigned short* wp = &ldsW[buf][(nt * 16 + mrow) * LDT + hi * 8];
      b.u[0] = *(const uint4*)(wp);
      b.u[1] = *(const uint4*)(wp + 16);
      acc[nt] = __builtin_amdgcn_wmma_f32_16x16x32_bf16(
          false, a.v, false, b.v, (short)0, acc[nt], false, false);
    }

    if (hasNext) STORE_CHUNK(buf ^ 1)      // convert + stash into other buffer
    buf ^= 1;
  }
#undef LOAD_CHUNK
#undef STORE_CHUNK

  // Bias + ReLU + store. C/D layout: VGPR r -> M = 8*hi + r, N = mrow.
  #pragma unroll
  for (int nt = 0; nt < 8; ++nt) {
    const int n = nt * 16 + mrow;
    const float bv = bias[n];
    #pragma unroll
    for (int r = 0; r < 8; ++r) {
      const int m = rowBase + wave * 16 + hi * 8 + r;
      float v = acc[nt][r] + bv;
      v = v > 0.0f ? v : 0.0f;
      dst[(size_t)m * HDIM + n] = v;
    }
  }
}

// ---------------------------------------------------------------------------
// Kernel 2: per-sample column means of ta/tb over the 1280 token rows.
// grid = 64 (32 samples x 2 tensors), 128 threads (one per feature dim).
// ---------------------------------------------------------------------------
__global__ __launch_bounds__(128)
void mean_kernel(const float* __restrict__ ta, const float* __restrict__ tb,
                 float* __restrict__ taMean,   float* __restrict__ tbMean)
{
  const int which = blockIdx.x & 1;
  const int b     = blockIdx.x >> 1;
  const float* src = which ? tb : ta;
  float*       dst = which ? tbMean : taMean;
  const int tid = threadIdx.x;
  const float* base = src + (size_t)b * (RV_NUM * RV_LEN) * HDIM + tid;
  float s = 0.0f;
  for (int m = 0; m < RV_NUM * RV_LEN; ++m) s += base[(size_t)m * HDIM];
  dst[b * HDIM + tid] = s * (1.0f / (RV_NUM * RV_LEN));
}

// ---------------------------------------------------------------------------
// Kernel 3: scores = (feat_row . other_mean)/1280, masked softmax over 128
// tokens, weighted feature sum. grid = 640 (320 per side), 128 threads.
// ---------------------------------------------------------------------------
__global__ __launch_bounds__(128)
void attn_kernel(const float* __restrict__ ta, const float* __restrict__ tb,
                 const float* __restrict__ taMean, const float* __restrict__ tbMean,
                 const int* __restrict__ maskA,    const int* __restrict__ maskB,
                 float* __restrict__ out)
{
  __shared__ float meanS[HDIM];
  __shared__ float red[RV_LEN];
  __shared__ float wS[RV_LEN];

  const int tid  = threadIdx.x;
  const int side = (blockIdx.x >= NREV) ? 1 : 0;
  const int n    = blockIdx.x - side * NREV;
  const int b    = n / RV_NUM;

  const float* feat = side ? tb     : ta;
  const float* om   = side ? taMean : tbMean;
  const int*   msk  = side ? maskB  : maskA;
  float* outF = out + (size_t)side * (NREV * HDIM);        // a_out | b_out
  float* outW = out + (size_t)(2 + side) * (NREV * HDIM);  // atob_w | btoa_w

  meanS[tid] = om[b * HDIM + tid];
  __syncthreads();

  // score for token l = tid
  const float* row = feat + ((size_t)n * RV_LEN + tid) * HDIM;
  float s = 0.0f;
  #pragma unroll 4
  for (int d = 0; d < HDIM; ++d) s = fmaf(row[d], meanS[d], s);
  s *= (1.0f / (RV_NUM * RV_LEN));

  float logit = (msk[n * RV_LEN + tid] > 0) ? s : -1e9f;

  // block max
  red[tid] = logit; __syncthreads();
  for (int off = 64; off > 0; off >>= 1) {
    if (tid < off) red[tid] = fmaxf(red[tid], red[tid + off]);
    __syncthreads();
  }
  const float mx = red[0];
  __syncthreads();

  // block sum of exp
  const float e = __expf(logit - mx);
  red[tid] = e; __syncthreads();
  for (int off = 64; off > 0; off >>= 1) {
    if (tid < off) red[tid] += red[tid + off];
    __syncthreads();
  }
  const float w = e / red[0];
  wS[tid] = w;
  outW[(size_t)n * RV_LEN + tid] = w;
  __syncthreads();

  // weighted sum over tokens: thread tid = output dim d (coalesced reads)
  const float* base = feat + (size_t)n * RV_LEN * HDIM;
  float acc = 0.0f;
  for (int l = 0; l < RV_LEN; ++l) acc = fmaf(wS[l], base[(size_t)l * HDIM + tid], acc);
  outF[(size_t)n * HDIM + tid] = acc;
}

// ---------------------------------------------------------------------------
extern "C" void kernel_launch(void* const* d_in, const int* in_sizes, int n_in,
                              void* d_out, int out_size, void* d_ws, size_t ws_size,
                              hipStream_t stream) {
  (void)in_sizes; (void)n_in; (void)out_size; (void)ws_size;
  const float* seqA = (const float*)d_in[0];
  const float* seqB = (const float*)d_in[1];
  const int*   mA   = (const int*)d_in[2];
  const int*   mB   = (const int*)d_in[3];
  const float* W    = (const float*)d_in[4];
  const float* bias = (const float*)d_in[5];
  float* out = (float*)d_out;

  // workspace: ta (20MB) | tb (20MB) | ta_mean | tb_mean  (all f32)
  float* ws     = (float*)d_ws;
  float* ta     = ws;
  float* tb     = ws + (size_t)ROWS * HDIM;
  float* taMean = tb + (size_t)ROWS * HDIM;
  float* tbMean = taMean + BZ * HDIM;

  dim3 g1(ROWS / 128, 2);
  fc_wmma_kernel<<<g1, 256, 0, stream>>>(seqA, seqB, W, bias, ta, tb);
  mean_kernel<<<BZ * 2, 128, 0, stream>>>(ta, tb, taMean, tbMean);
  attn_kernel<<<2 * NREV, 128, 0, stream>>>(ta, tb, taMean, tbMean, mA, mB, out);
}
